// RecoGraphCore_65300682768558
// MI455X (gfx1250) — matrix operand order repaired and checked by reading, not compiled, not run on hardware
//
#include <hip/hip_runtime.h>

// ---------------------------------------------------------------------------
// RecoGraphCore forward for MI455X (gfx1250, wave32).
// Heavy GEMM (dots: 2112x32768, K=128) runs on v_wmma_f32_16x16x32_bf16.
// Everything else is memory-bound or tiny -> fp32 VALU kernels.
// ---------------------------------------------------------------------------

typedef __attribute__((ext_vector_type(16))) __bf16 bf16x16;
typedef __attribute__((ext_vector_type(8)))  __bf16 bf16x8;
typedef __attribute__((ext_vector_type(8)))  float  f32x8;

#define NNODES   2048
#define RR       16
#define SDIM     64
#define MDIM     128
#define VOC      256
#define OMM      64
#define NCLS     10
#define NR       32768              // NNODES*RR
#define MS       2112               // OMM + NNODES
#define IB       85                 // (2*SD+MD)/3
#define IC       106                // (SD+2*MD)/3

// ---- workspace layout (float units, everything 64-float = 256B aligned) ----
#define OFF_ACC   ((size_t)0)                         // 7 (sum,sumsq) pairs
#define OFF_FIN   ((size_t)64)                        // 7 (mean,inv) pairs
#define OFF_VB1   ((size_t)256)                       // vision ping (<=64*26*26)
#define OFF_VB2   (OFF_VB1 + 50432)                   // vision pong
#define OFF_MIN   (OFF_VB2 + 50432)                   // memes_in raw (64*128)
#define OFF_Y1    (OFF_MIN + 8192)                    // 32768*85
#define OFF_Y2    (OFF_Y1 + (size_t)NR*IB)            // 32768*106
#define OFF_Y3    (OFF_Y2 + (size_t)NR*IC)            // 32768*128
#define OFF_RECVF (OFF_Y3 + (size_t)NR*MDIM)          // recv fp32 normalized
#define OFF_RECVB (OFF_RECVF + (size_t)NR*MDIM)       // recv bf16 (2 per float)
#define OFF_SENDB (OFF_RECVB + (size_t)NR*MDIM/2)     // senders bf16
#define OFF_DOTS  (OFF_SENDB + (size_t)MS*MDIM/2)     // 2112*32768 fp32
#define OFF_F     (OFF_DOTS + (size_t)MS*NR)          // f colsums (32768)
#define OFF_MEMES (OFF_F + NR)                        // 2048*128
#define OFF_MEM2  (OFF_MEMES + (size_t)NNODES*MDIM)   // 2048*128
#define OFF_T1    (OFF_MEM2 + (size_t)NNODES*MDIM)    // 2048*64
#define OFF_T2    (OFF_T1 + (size_t)NNODES*64)        // 2048*32
#define OFF_WTS   (OFF_T2 + (size_t)NNODES*32)        // 2048
#define OFF_RSINV (OFF_WTS + 2048)                    // 2112 row-sum inverses

// stat slots: 0 visionFC, 1..3 inception L1..L3, 4..6 classifier t1/t2/wts
#define S_VFC 0
#define S_I1  1
#define S_I2  2
#define S_I3  3
#define S_CA  4
#define S_CB  5
#define S_CC  6

// ---------------------------------------------------------------------------
static __device__ __forceinline__ float blockReduceSum(float v, float* sh) {
    int tid = threadIdx.x;
    __syncthreads();
    sh[tid] = v;
    __syncthreads();
    for (int s = blockDim.x >> 1; s > 0; s >>= 1) {
        if (tid < s) sh[tid] += sh[tid + s];
        __syncthreads();
    }
    return sh[0];
}

__global__ void zero_kernel(float* p, int n) {
    int i = blockIdx.x * blockDim.x + threadIdx.x;
    if (i < n) p[i] = 0.f;
}

// ---------------------------------------------------------------------------
// Single-workgroup conv3x3 + whole-tensor inorm (+relu / +2x2 maxpool+relu).
// Whole output (<=43K elems) handled by one 1024-thread block so the global
// mean/std reduction stays in LDS.
// ---------------------------------------------------------------------------
__global__ void conv_block(const float* __restrict__ inp,
                           const float* __restrict__ wgt,
                           const float* __restrict__ bias,
                           float* tmp, float* outp,
                           int Cin, int Cout, int H, int W,
                           int pad, int doPool, int doRelu) {
    __shared__ float red[1024];
    __shared__ float s_mean, s_inv;
    const int Hout = H + 2 * pad - 2, Wout = W + 2 * pad - 2;
    const int total = Cout * Hout * Wout;
    float ls = 0.f, lq = 0.f;
    for (int idx = threadIdx.x; idx < total; idx += blockDim.x) {
        int co = idx / (Hout * Wout);
        int rem = idx % (Hout * Wout);
        int oy = rem / Wout, ox = rem % Wout;
        float acc = bias[co];
        for (int ci = 0; ci < Cin; ++ci) {
            const float* ip = inp + (size_t)ci * H * W;
            const float* wp = wgt + ((size_t)co * Cin + ci) * 9;
            for (int ky = 0; ky < 3; ++ky) {
                int iy = oy - pad + ky;
                if (iy < 0 || iy >= H) continue;
                for (int kx = 0; kx < 3; ++kx) {
                    int ix = ox - pad + kx;
                    if (ix < 0 || ix >= W) continue;
                    acc += ip[iy * W + ix] * wp[ky * 3 + kx];
                }
            }
        }
        tmp[idx] = acc; ls += acc; lq += acc * acc;
    }
    float ts = blockReduceSum(ls, red);
    float tq = blockReduceSum(lq, red);
    if (threadIdx.x == 0) {
        float mean = ts / (float)total;
        float var = (tq - (float)total * mean * mean) / (float)(total - 1);
        s_mean = mean; s_inv = 1.f / (sqrtf(fmaxf(var, 0.f)) + 1e-3f);
    }
    __syncthreads();
    for (int idx = threadIdx.x; idx < total; idx += blockDim.x) {
        float v = (tmp[idx] - s_mean) * s_inv;
        if (doRelu && !doPool) v = fmaxf(v, 0.f);
        tmp[idx] = v;
    }
    __syncthreads();
    if (doPool) {
        int Hp = Hout / 2, Wp = Wout / 2, tp = Cout * Hp * Wp;
        for (int idx = threadIdx.x; idx < tp; idx += blockDim.x) {
            int co = idx / (Hp * Wp);
            int rem = idx % (Hp * Wp);
            int py = rem / Wp, px = rem % Wp;
            const float* base = tmp + (size_t)co * Hout * Wout;
            float a = base[(2 * py) * Wout + 2 * px];
            float b = base[(2 * py) * Wout + 2 * px + 1];
            float c = base[(2 * py + 1) * Wout + 2 * px];
            float d = base[(2 * py + 1) * Wout + 2 * px + 1];
            float m = fmaxf(fmaxf(a, b), fmaxf(c, d));
            outp[idx] = fmaxf(m, 0.f);   // relu(pool(.))
        }
    }
}

// ---------------------------------------------------------------------------
// Generic Y = (norm(relu?) X) @ W + b ; accumulates whole-tensor (sum,sumsq)
// into statsOut via one atomicAdd pair per block.
// ---------------------------------------------------------------------------
__global__ void linear_stats(const float* __restrict__ X, int ldx,
                             const float* __restrict__ statsIn, int reluIn,
                             const float* __restrict__ W,
                             const float* __restrict__ B,
                             float* __restrict__ Y, float* statsOut,
                             int rows, int K, int Ncols) {
    __shared__ float red[256];
    int idx = blockIdx.x * blockDim.x + threadIdx.x;
    float val = 0.f;
    int valid = idx < rows * Ncols;
    if (valid) {
        int r = idx / Ncols, c = idx % Ncols;
        float mean = 0.f, inv = 1.f;
        if (statsIn) { mean = statsIn[0]; inv = statsIn[1]; }
        float acc = B ? B[c] : 0.f;
        const float* xr = X + (size_t)r * ldx;
        for (int k = 0; k < K; ++k) {
            float xv = xr[k];
            if (statsIn) xv = (xv - mean) * inv;
            if (reluIn) xv = fmaxf(xv, 0.f);
            acc += xv * W[(size_t)k * Ncols + c];
        }
        Y[(size_t)r * Ncols + c] = acc;
        val = acc;
    }
    float s = blockReduceSum(valid ? val : 0.f, red);
    float q = blockReduceSum(valid ? val * val : 0.f, red);
    if (threadIdx.x == 0 && statsOut) {
        atomicAdd(&statsOut[0], s);
        atomicAdd(&statsOut[1], q);
    }
}

__global__ void finalize_stats(const float* acc, float* fin, float count) {
    if (threadIdx.x == 0) {
        float mean = acc[0] / count;
        float var = (acc[1] - count * mean * mean) / (count - 1.f);
        fin[0] = mean;
        fin[1] = 1.f / (sqrtf(fmaxf(var, 0.f)) + 1e-3f);
    }
}

// Apply final inorm to Y3 -> recv fp32 (kept for weighted sums) + bf16 (WMMA B)
__global__ void recv_finalize(const float* __restrict__ Y3,
                              const float* __restrict__ st,
                              float* __restrict__ recvF,
                              __bf16* __restrict__ recvB, int n) {
    int i = blockIdx.x * blockDim.x + threadIdx.x;
    if (i >= n) return;
    float v = (Y3[i] - st[0]) * st[1];
    recvF[i] = v;
    recvB[i] = (__bf16)v;
}

// senders = concat(inorm(memes_in_raw), memes_last) -> bf16 (WMMA A)
__global__ void senders_pack(const float* __restrict__ memesInRaw,
                             const float* __restrict__ st,
                             const float* __restrict__ memesLast,
                             __bf16* __restrict__ outB) {
    int i = blockIdx.x * blockDim.x + threadIdx.x;
    if (i >= MS * MDIM) return;
    int r = i >> 7;
    float v;
    if (r < OMM) v = (memesInRaw[i] - st[0]) * st[1];
    else         v = memesLast[(size_t)(r - OMM) * MDIM + (i & 127)];
    outB[i] = (__bf16)v;
}

// ---------------------------------------------------------------------------
// dots[m, nr] = <senders[m,:], recv[nr,:]>  (M=2112, N=32768, K=128)
// bf16 WMMA 16x16x32; wave computes a 16x64 tile (4 accumulators), K unrolled
// 4x -> 16 v_wmma per wave iteration. Block = 8 waves = 16x512 tile.
// A layout per lane: row=lane&15, 16 contiguous K starting at kb + (lane>=16)*16.
// B per lane: col=lane&15,
// 16 contiguous K starting at kb + (lane>=16)*16 (contiguous in row-major recv).
// ---------------------------------------------------------------------------
__global__ void dots_wmma(const __bf16* __restrict__ S,
                          const __bf16* __restrict__ Rv,
                          float* __restrict__ D) {
    const int lane = threadIdx.x & 31;
    const int wave = threadIdx.x >> 5;
    const int l15 = lane & 15;
    const int half = lane >> 4;
    const int row16 = blockIdx.y * 16;
    const int colBase = blockIdx.x * 512 + wave * 64;
    const int m = row16 + l15;

    f32x8 acc[4];
    acc[0] = (f32x8){}; acc[1] = (f32x8){}; acc[2] = (f32x8){}; acc[3] = (f32x8){};

#pragma unroll
    for (int kt = 0; kt < 4; ++kt) {
        const int kb = kt * 32;
        const __bf16* ap = S + (size_t)m * MDIM + kb + half * 16;
        bf16x8 alo = *(const bf16x8*)(ap);
        bf16x8 ahi = *(const bf16x8*)(ap + 8);
        bf16x16 a = __builtin_shufflevector(alo, ahi,
            0, 1, 2, 3, 4, 5, 6, 7, 8, 9, 10, 11, 12, 13, 14, 15);
#pragma unroll
        for (int ct = 0; ct < 4; ++ct) {
            const __bf16* bp = Rv + (size_t)(colBase + ct * 16 + l15) * MDIM
                                  + kb + half * 16;
            bf16x16 b = *(const bf16x16*)(bp);
            acc[ct] = __builtin_amdgcn_wmma_f32_16x16x32_bf16(
                false, a, false, b, (short)0, acc[ct], false, false);
        }
    }

    const int rbase = row16 + half * 8;
#pragma unroll
    for (int ct = 0; ct < 4; ++ct) {
        float* dp = D + (size_t)rbase * NR + (size_t)colBase + ct * 16 + l15;
#pragma unroll
        for (int j = 0; j < 8; ++j)
            dp[(size_t)j * NR] = acc[ct][j];
    }
}

// ---------------------------------------------------------------------------
// fracs on dots rows (len 32768): standardize (ddof=1), tanh(h/sc)*sc,
// sh^h = exp(h*log_sh), write transformed in place, store 1/(rowsum+eps).
// One workgroup per row; two streaming passes (row stays hot in L2).
// ---------------------------------------------------------------------------
__global__ void frac_rows(float* __restrict__ mat, int cols,
                          const float* __restrict__ scp,
                          const float* __restrict__ shp,
                          float* __restrict__ rsinv) {
    __shared__ float red[1024];
    __shared__ float bm, bi;
    const int tid = threadIdx.x;
    float* mp = mat + (size_t)blockIdx.x * cols;
    float ls = 0.f, lq = 0.f;
    for (int c = tid; c < cols; c += blockDim.x) {
        float v = mp[c]; ls += v; lq += v * v;
    }
    float s = blockReduceSum(ls, red);
    float q = blockReduceSum(lq, red);
    if (tid == 0) {
        float mean = s / (float)cols;
        float var = (q - (float)cols * mean * mean) / (float)(cols - 1);
        bm = mean; bi = 1.f / (sqrtf(fmaxf(var, 0.f)) + 1e-3f);
    }
    __syncthreads();
    const float sc = expf(scp[0]);     // sc = exp(g_sc)
    const float lnsh = shp[0];         // log(sh) = g_sh (sh = exp(g_sh))
    float lr = 0.f;
    for (int c = tid; c < cols; c += blockDim.x) {
        float h = (mp[c] - bm) * bi;
        h = tanhf(h / sc) * sc;
        h = expf(h * lnsh);
        mp[c] = h; lr += h;
    }
    float rs = blockReduceSum(lr, red);
    if (tid == 0) rsinv[blockIdx.x] = 1.f / (rs + 1e-3f);
}

// f[c] = sum_r mat[r][c] * rsinv[r]   (folds the fraction divide -> saves a pass)
__global__ void col_sum_scaled(const float* __restrict__ mat,
                               const float* __restrict__ rsinv,
                               float* __restrict__ f, int rows, int cols) {
    int c = blockIdx.x * blockDim.x + threadIdx.x;
    if (c >= cols) return;
    float s = 0.f;
    for (int r = 0; r < rows; ++r)
        s += mat[(size_t)r * cols + c] * rsinv[r];
    f[c] = s;
}

// memes[n][e] = sum_r recvF[n*16+r][e] * f[n*16+r]
__global__ void memes_combine(const float* __restrict__ recvF,
                              const float* __restrict__ f,
                              float* __restrict__ memes) {
    int i = blockIdx.x * blockDim.x + threadIdx.x;
    if (i >= NNODES * MDIM) return;
    int n = i >> 7, e = i & 127;
    float s = 0.f;
    for (int r = 0; r < RR; ++r)
        s += recvF[(size_t)(n * RR + r) * MDIM + e] * f[n * RR + r];
    memes[i] = s;
}

// ---------------------------------------------------------------------------
// Reconstructor: per node, w = memes @ vocab[n]; fracs over 256; reconstruct.
// Fully fused per workgroup so the 268MB vocab streams once per matvec.
// ---------------------------------------------------------------------------
__global__ void recon(const float* __restrict__ memes,
                      const float* __restrict__ vocab,
                      const float* __restrict__ scp,
                      const float* __restrict__ shp,
                      float* __restrict__ memes2) {
    __shared__ float red[256];
    __shared__ float me[MDIM];
    __shared__ float wv[VOC];
    __shared__ float bm, bi, brs;
    const int n = blockIdx.x, tid = threadIdx.x;
    if (tid < MDIM) me[tid] = memes[(size_t)n * MDIM + tid];
    __syncthreads();
    const float* vb = vocab + (size_t)n * MDIM * VOC;
    float acc = 0.f;
    for (int e = 0; e < MDIM; ++e)
        acc += me[e] * vb[(size_t)e * VOC + tid];
    float s = blockReduceSum(acc, red);
    float q = blockReduceSum(acc * acc, red);
    if (tid == 0) {
        float mean = s / (float)VOC;
        float var = (q - (float)VOC * mean * mean) / (float)(VOC - 1);
        bm = mean; bi = 1.f / (sqrtf(fmaxf(var, 0.f)) + 1e-3f);
    }
    __syncthreads();
    const float sc = expf(scp[0]);
    const float lnsh = shp[0];
    float h = (acc - bm) * bi;
    h = tanhf(h / sc) * sc;
    h = expf(h * lnsh);
    float rs = blockReduceSum(h, red);
    if (tid == 0) brs = 1.f / (rs + 1e-3f);
    __syncthreads();
    wv[tid] = h * brs;
    __syncthreads();
    if (tid < MDIM) {
        float s2 = 0.f;
        const float* vr = vb + (size_t)tid * VOC;
        for (int v = 0; v < VOC; ++v) s2 += vr[v] * wv[v];
        memes2[(size_t)n * MDIM + tid] = s2;
    }
}

// ---------------------------------------------------------------------------
// Classifier tail: g = memes2^T wts -> standardize -> 128->256->512->10
// (tiny: single 512-thread workgroup with LDS staging)
// ---------------------------------------------------------------------------
__global__ void classifier_tail(const float* __restrict__ memes2,
                                const float* __restrict__ wtsRaw,
                                const float* __restrict__ wst,
                                const float* cw1, const float* cb1,
                                const float* cw2, const float* cb2,
                                const float* cw3, const float* cb3,
                                float* __restrict__ out) {
    __shared__ float red[512];
    __shared__ float gpart[512];
    __shared__ float gvec[128], u1[256], u2[512];
    __shared__ float bm, bi;
    const int tid = threadIdx.x;
    const float wm = wst[0], wi = wst[1];
    // g[e] = sum_n memes2[n][e] * inorm(wts)[n]
    int e = tid & 127, part = tid >> 7;     // 4 partitions of 512 rows
    float p = 0.f;
    for (int n = part * 512; n < (part + 1) * 512; ++n)
        p += memes2[(size_t)n * MDIM + e] * ((wtsRaw[n] - wm) * wi);
    gpart[tid] = p;
    __syncthreads();
    if (tid < 128) gvec[tid] = gpart[tid] + gpart[tid + 128] +
                               gpart[tid + 256] + gpart[tid + 384];
    __syncthreads();
    // standardize g (128, ddof=1)
    float v = (tid < 128) ? gvec[tid] : 0.f;
    float s = blockReduceSum(v, red);
    float q = blockReduceSum(v * v, red);
    if (tid == 0) {
        float mean = s / 128.f;
        float var = (q - 128.f * mean * mean) / 127.f;
        bm = mean; bi = 1.f / (sqrtf(fmaxf(var, 0.f)) + 1e-3f);
    }
    __syncthreads();
    if (tid < 128) gvec[tid] = (gvec[tid] - bm) * bi;
    __syncthreads();
    // u1 = relu(inorm(g @ cw1 + cb1)), 256
    float a1 = 0.f;
    if (tid < 256) {
        a1 = cb1[tid];
        for (int k = 0; k < 128; ++k) a1 += gvec[k] * cw1[k * 256 + tid];
    }
    s = blockReduceSum(tid < 256 ? a1 : 0.f, red);
    q = blockReduceSum(tid < 256 ? a1 * a1 : 0.f, red);
    if (tid == 0) {
        float mean = s / 256.f;
        float var = (q - 256.f * mean * mean) / 255.f;
        bm = mean; bi = 1.f / (sqrtf(fmaxf(var, 0.f)) + 1e-3f);
    }
    __syncthreads();
    if (tid < 256) u1[tid] = fmaxf((a1 - bm) * bi, 0.f);
    __syncthreads();
    // u2 = relu(inorm(u1 @ cw2 + cb2)), 512
    float a2 = cb2[tid];
    for (int k = 0; k < 256; ++k) a2 += u1[k] * cw2[k * 512 + tid];
    s = blockReduceSum(a2, red);
    q = blockReduceSum(a2 * a2, red);
    if (tid == 0) {
        float mean = s / 512.f;
        float var = (q - 512.f * mean * mean) / 511.f;
        bm = mean; bi = 1.f / (sqrtf(fmaxf(var, 0.f)) + 1e-3f);
    }
    __syncthreads();
    u2[tid] = fmaxf((a2 - bm) * bi, 0.f);
    __syncthreads();
    if (tid < NCLS) {
        float o = cb3[tid];
        for (int k = 0; k < 512; ++k) o += u2[k] * cw3[k * NCLS + tid];
        out[tid] = o;
    }
}

// ---------------------------------------------------------------------------
extern "C" void kernel_launch(void* const* d_in, const int* in_sizes, int n_in,
                              void* d_out, int out_size, void* d_ws, size_t ws_size,
                              hipStream_t stream) {
    (void)in_sizes; (void)n_in; (void)out_size; (void)ws_size;
    const float* x          = (const float*)d_in[0];
    const float* memes_last = (const float*)d_in[1];
    const float* seeds      = (const float*)d_in[2];
    const float* vw[6]; const float* vb[6];
    for (int i = 0; i < 6; ++i) {
        vw[i] = (const float*)d_in[3 + 2 * i];
        vb[i] = (const float*)d_in[4 + 2 * i];
    }
    const float* vlw  = (const float*)d_in[15];
    const float* vlb  = (const float*)d_in[16];
    const float* iw1  = (const float*)d_in[17];
    const float* ib1  = (const float*)d_in[18];
    const float* iw2  = (const float*)d_in[19];
    const float* ib2  = (const float*)d_in[20];
    const float* iw3  = (const float*)d_in[21];
    const float* ib3  = (const float*)d_in[22];
    const float* g_sc = (const float*)d_in[23];
    const float* g_sh = (const float*)d_in[24];
    const float* voc  = (const float*)d_in[25];
    const float* r_sc = (const float*)d_in[26];
    const float* r_sh = (const float*)d_in[27];
    const float* mw1  = (const float*)d_in[28];
    const float* mb1  = (const float*)d_in[29];
    const float* mw2  = (const float*)d_in[30];
    const float* mb2  = (const float*)d_in[31];
    const float* mw3  = (const float*)d_in[32];
    const float* mb3  = (const float*)d_in[33];
    const float* cw1  = (const float*)d_in[34];
    const float* cb1  = (const float*)d_in[35];
    const float* cw2  = (const float*)d_in[36];
    const float* cb2  = (const float*)d_in[37];
    const float* cw3  = (const float*)d_in[38];
    const float* cb3  = (const float*)d_in[39];

    float* ws   = (float*)d_ws;
    float* ACC  = ws + OFF_ACC;
    float* FIN  = ws + OFF_FIN;
    float* B1   = ws + OFF_VB1;
    float* B2   = ws + OFF_VB2;
    float* MINR = ws + OFF_MIN;
    float* Y1   = ws + OFF_Y1;
    float* Y2   = ws + OFF_Y2;
    float* Y3   = ws + OFF_Y3;
    float* RECF = ws + OFF_RECVF;
    __bf16* RECB = (__bf16*)(ws + OFF_RECVB);
    __bf16* SNDB = (__bf16*)(ws + OFF_SENDB);
    float* DOTS = ws + OFF_DOTS;
    float* FCOL = ws + OFF_F;
    float* MEM  = ws + OFF_MEMES;
    float* MEM2 = ws + OFF_MEM2;
    float* T1   = ws + OFF_T1;
    float* T2   = ws + OFF_T2;
    float* WTSR = ws + OFF_WTS;
    float* RSI  = ws + OFF_RSINV;

    // 0) zero stats accumulators (deterministic per-launch)
    zero_kernel<<<1, 128, 0, stream>>>(ws, 128);

    // 1) vision CNN (single-workgroup layers; inorm stats in LDS)
    conv_block<<<1, 1024, 0, stream>>>(x,  vw[0], vb[0], B1, B1, 1,  64, 28, 28, 0, 0, 1);
    conv_block<<<1, 1024, 0, stream>>>(B1, vw[1], vb[1], B2, B1, 64, 64, 26, 26, 0, 1, 1);
    conv_block<<<1, 1024, 0, stream>>>(B1, vw[2], vb[2], B2, B2, 64, 64, 12, 12, 0, 0, 1);
    conv_block<<<1, 1024, 0, stream>>>(B2, vw[3], vb[3], B1, B2, 64, 64, 10, 10, 0, 1, 1);
    conv_block<<<1, 1024, 0, stream>>>(B2, vw[4], vb[4], B1, B1, 64, 64, 4, 4, 1, 0, 1);
    conv_block<<<1, 1024, 0, stream>>>(B1, vw[5], vb[5], B2, B2, 64, 64, 4, 4, 1, 0, 0);
    // vision FC: relu(flat) @ vlw + vlb  (inorm stats -> S_VFC)
    linear_stats<<<32, 256, 0, stream>>>(B2, 1024, nullptr, 1, vlw, vlb,
                                         MINR, ACC + 2 * S_VFC, 1, 1024, 8192);
    finalize_stats<<<1, 32, 0, stream>>>(ACC + 2 * S_VFC, FIN + 2 * S_VFC, 8192.f);

    // 2) inception MLP on seeds (32768 rows)
    linear_stats<<<(NR * IB + 255) / 256, 256, 0, stream>>>(
        seeds, SDIM, nullptr, 0, iw1, ib1, Y1, ACC + 2 * S_I1, NR, SDIM, IB);
    finalize_stats<<<1, 32, 0, stream>>>(ACC + 2 * S_I1, FIN + 2 * S_I1, (float)NR * IB);
    linear_stats<<<(NR * IC + 255) / 256, 256, 0, stream>>>(
        Y1, IB, FIN + 2 * S_I1, 1, iw2, ib2, Y2, ACC + 2 * S_I2, NR, IB, IC);
    finalize_stats<<<1, 32, 0, stream>>>(ACC + 2 * S_I2, FIN + 2 * S_I2, (float)NR * IC);
    linear_stats<<<(NR * MDIM + 255) / 256, 256, 0, stream>>>(
        Y2, IC, FIN + 2 * S_I2, 1, iw3, ib3, Y3, ACC + 2 * S_I3, NR, IC, MDIM);
    finalize_stats<<<1, 32, 0, stream>>>(ACC + 2 * S_I3, FIN + 2 * S_I3, (float)NR * MDIM);
    recv_finalize<<<(NR * MDIM + 255) / 256, 256, 0, stream>>>(
        Y3, FIN + 2 * S_I3, RECF, RECB, NR * MDIM);

    // 3) pack senders -> bf16, run the WMMA GEMM (dots)
    senders_pack<<<(MS * MDIM + 255) / 256, 256, 0, stream>>>(
        MINR, FIN + 2 * S_VFC, memes_last, SNDB);
    dim3 gDots(NR / 512, MS / 16, 1);
    dots_wmma<<<gDots, 256, 0, stream>>>(SNDB, RECB, DOTS);

    // 4) fracs over dots rows, column-sum -> f, weighted recv sum -> memes
    frac_rows<<<MS, 1024, 0, stream>>>(DOTS, NR, g_sc, g_sh, RSI);
    col_sum_scaled<<<NR / 256, 256, 0, stream>>>(DOTS, RSI, FCOL, MS, NR);
    memes_combine<<<(NNODES * MDIM + 255) / 256, 256, 0, stream>>>(RECF, FCOL, MEM);

    // 5) reconstructor remix (fused per node)
    recon<<<NNODES, 256, 0, stream>>>(MEM, voc, r_sc, r_sh, MEM2);

    // 6) classifier weight path
    linear_stats<<<(NNODES * 64 + 255) / 256, 256, 0, stream>>>(
        MEM2, MDIM, nullptr, 0, mw1, mb1, T1, ACC + 2 * S_CA, NNODES, MDIM, 64);
    finalize_stats<<<1, 32, 0, stream>>>(ACC + 2 * S_CA, FIN + 2 * S_CA, (float)NNODES * 64);
    linear_stats<<<(NNODES * 32 + 255) / 256, 256, 0, stream>>>(
        T1, 64, FIN + 2 * S_CA, 1, mw2, mb2, T2, ACC + 2 * S_CB, NNODES, 64, 32);
    finalize_stats<<<1, 32, 0, stream>>>(ACC + 2 * S_CB, FIN + 2 * S_CB, (float)NNODES * 32);
    linear_stats<<<(NNODES + 255) / 256, 256, 0, stream>>>(
        T2, 32, FIN + 2 * S_CB, 1, mw3, mb3, WTSR, ACC + 2 * S_CC, NNODES, 32, 1);
    finalize_stats<<<1, 32, 0, stream>>>(ACC + 2 * S_CC, FIN + 2 * S_CC, (float)NNODES);

    // 7) classifier tail -> (10,)
    classifier_tail<<<1, 512, 0, stream>>>(MEM2, WTSR, FIN + 2 * S_CC,
                                           cw1, cb1, cw2, cb2, cw3, cb3,
                                           (float*)d_out);
}
